// CronRootAttention_50929722196093
// MI455X (gfx1250) — matrix operand order repaired and checked by reading, not compiled, not run on hardware
//
#include <hip/hip_runtime.h>
#include <hip/hip_bf16.h>

// ---------------------------------------------------------------------------
// CronRootAttention for MI455X (gfx1250, wave32, WMMA).
// Pipeline: f32->bf16 convert, Q/K/V GEMMs (v_wmma_f32_16x16x32_bf16 with
// 64x32 macro-tiles per wave for 1.5 B128-loads/WMMA), sparse local+strided
// attention fully on WMMA (scores computed transposed so softmax is in-lane;
// P^T reused directly as the WMMA B operand), final Wo GEMM.
// ---------------------------------------------------------------------------

typedef __attribute__((ext_vector_type(16))) __bf16       v16bf;
typedef __attribute__((ext_vector_type(8)))  float        v8f;
typedef __attribute__((ext_vector_type(4)))  unsigned int u32x4;
typedef __attribute__((ext_vector_type(2)))  __bf16       bf16x2;

union FragBF { u32x4 q[2]; v16bf v; };

#define S_LEN  2048
#define DMODEL 1024
#define NH     16
#define NKV    4
#define HD     64
#define KVCOLS (NKV * HD)   // 256
#define WWIN   46           // ceil(sqrt(2048))
#define NSTR   44           // # strided summary keys
#define NEGINF (-1.0e9f)

// ---------------------------------------------------------------------------
// f32 -> bf16 conversion, 4 elements / thread
// ---------------------------------------------------------------------------
__global__ void __launch_bounds__(256)
cvt_bf16_kernel(const float* __restrict__ in, __bf16* __restrict__ out, int n4) {
    int i = blockIdx.x * blockDim.x + threadIdx.x;
    if (i < n4) {
        float4 f = ((const float4*)in)[i];
        bf16x2 a, b;
        a[0] = (__bf16)f.x; a[1] = (__bf16)f.y;
        b[0] = (__bf16)f.z; b[1] = (__bf16)f.w;
        ((bf16x2*)out)[2 * i]     = a;
        ((bf16x2*)out)[2 * i + 1] = b;
    }
}

// ---------------------------------------------------------------------------
// Generic  out[s,o] = sum_i A[s,i] * W[o,i]   (A: MxK bf16, W: NxK bf16)
// 64x32 macro-tile per wave: 4 M-tiles x 2 N-tiles, 8 f32 accumulators.
// Per 32-K chunk: 12 B128 loads feed 8 WMMAs (A frags reused across N,
// B frags reused across M).  Requires M%64==0, N%32==0, K%32==0.
// ---------------------------------------------------------------------------
__global__ void __launch_bounds__(256)
gemm_bf16_wmma(const __bf16* __restrict__ A, const __bf16* __restrict__ W,
               float* __restrict__ outF, __bf16* __restrict__ outB,
               int M, int N, int K) {
    const int lane = threadIdx.x & 31;
    const int wid  = threadIdx.x >> 5;
    const int gw   = blockIdx.x * (blockDim.x >> 5) + wid;
    const int mb   = M >> 6, nb = N >> 5;           // 64-row / 32-col blocks
    if (gw >= mb * nb) return;                      // wave-uniform
    const int bm  = gw % mb, bn = gw / mb;
    const int n16 = lane & 15, g = lane >> 4;

    long arow[4], wrow[2];
#pragma unroll
    for (int i = 0; i < 4; ++i)                     // A lane rows (m = lane&15)
        arow[i] = (long)(bm * 64 + i * 16 + n16) * K;
#pragma unroll
    for (int j = 0; j < 2; ++j)                     // W lane rows (n = lane&15)
        wrow[j] = (long)(bn * 32 + j * 16 + n16) * K;

    v8f c[4][2];
#pragma unroll
    for (int i = 0; i < 4; ++i)
#pragma unroll
        for (int j = 0; j < 2; ++j) c[i][j] = (v8f){};

    for (int kc = 0; kc < K; kc += 32) {
        __builtin_prefetch(A + arow[0] + kc + 256, 0, 1);   // global_prefetch_b8
        FragBF a[4], b[2];
#pragma unroll
        for (int i = 0; i < 4; ++i) {
            // A 16x32 frag: elems 0..7 -> k=g*8.. ; 8..15 -> k=16+g*8..
            a[i].q[0] = *(const u32x4*)(A + arow[i] + kc + g * 8);
            a[i].q[1] = *(const u32x4*)(A + arow[i] + kc + 16 + g * 8);
        }
#pragma unroll
        for (int j = 0; j < 2; ++j) {
            // B 32x16 frag: elems 0..7 -> k=g*8.. ; 8..15 -> k=16+g*8..
            b[j].q[0] = *(const u32x4*)(W + wrow[j] + kc + g * 8);
            b[j].q[1] = *(const u32x4*)(W + wrow[j] + kc + 16 + g * 8);
        }
#pragma unroll
        for (int i = 0; i < 4; ++i)
#pragma unroll
            for (int j = 0; j < 2; ++j)
                c[i][j] = __builtin_amdgcn_wmma_f32_16x16x32_bf16(
                    false, a[i].v, false, b[j].v, (short)0, c[i][j],
                    false, false);
    }

    // C/D layout: lane col n = n16 (feature), rows m = 8g+v
#pragma unroll
    for (int i = 0; i < 4; ++i)
#pragma unroll
        for (int j = 0; j < 2; ++j) {
            const int o = bn * 32 + j * 16 + n16;
            if (outF) {
#pragma unroll
                for (int v = 0; v < 8; ++v)
                    outF[(long)(bm * 64 + i * 16 + 8 * g + v) * N + o] = c[i][j][v];
            } else {
#pragma unroll
                for (int v = 0; v < 8; ++v)
                    outB[(long)(bm * 64 + i * 16 + 8 * g + v) * N + o] =
                        (__bf16)c[i][j][v];
            }
        }
}

// ---------------------------------------------------------------------------
// Sparse attention. One wave per (16-query block, head).
// Key slots: 0..63  = local tiles covering [q0-48, q0+15]
//            64..127 = strided keys pos = 45 + 46*j (44 valid)
// Scores computed transposed (S^T = Ktile * Q^T) so softmax is in-lane and
// P^T feeds the O^T = V^T * P^T WMMA directly after an xor-16 repack.
// ---------------------------------------------------------------------------
__global__ void __launch_bounds__(256)
attn_kernel(const __bf16* __restrict__ Qb, const __bf16* __restrict__ Kb,
            const __bf16* __restrict__ Vb, __bf16* __restrict__ Ob) {
    const int lane = threadIdx.x & 31;
    const int wid  = threadIdx.x >> 5;
    const int gw   = blockIdx.x * 8 + wid;
    const int h    = gw & (NH - 1);
    const int q0   = (gw >> 4) * 16;
    const int hkv  = h >> 2;
    const int n16  = lane & 15, g = lane >> 4;
    const int q    = q0 + n16;                  // this lane's query column

    // Q^T B-fragments, 2 chunks over head dim 64 (reused for all 8 key tiles)
    FragBF qf[2];
    {
        const __bf16* qrow = Qb + (long)q * DMODEL + h * HD;
#pragma unroll
        for (int c = 0; c < 2; ++c) {
            qf[c].q[0] = *(const u32x4*)(qrow + c * 32 + g * 8);
            qf[c].q[1] = *(const u32x4*)(qrow + c * 32 + 16 + g * 8);
        }
    }

    // ---- scores (transposed): p[t] rows = keys (8g+v), col = query ----
    v8f p[8];
    const float scale = 0.125f;                 // 1/sqrt(64)
#pragma unroll
    for (int t = 0; t < 8; ++t) {
        // A-matrix row m = lane&15 is a key slot of this tile
        int slot = t * 16 + n16;
        int pos  = (t < 4) ? (q0 - 48 + slot) : (45 + 46 * (slot - 64));
        int posc = pos < 0 ? 0 : (pos > S_LEN - 1 ? S_LEN - 1 : pos);
        const __bf16* krow = Kb + (long)posc * KVCOLS + hkv * HD;
        v8f c = {};
#pragma unroll
        for (int cc = 0; cc < 2; ++cc) {
            FragBF a;
            a.q[0] = *(const u32x4*)(krow + cc * 32 + g * 8);
            a.q[1] = *(const u32x4*)(krow + cc * 32 + 16 + g * 8);
            c = __builtin_amdgcn_wmma_f32_16x16x32_bf16(false, a.v, false,
                                                        qf[cc].v, (short)0, c,
                                                        false, false);
        }
        // mask (rows m = 8g+v are keys, column is my query q)
#pragma unroll
        for (int v = 0; v < 8; ++v) {
            int  ks = t * 16 + 8 * g + v;
            bool valid;
            if (t < 4) {
                int kpos = q0 - 48 + ks;
                valid = (kpos >= 0) && (kpos >= q - (WWIN - 1)) && (kpos <= q);
            } else {
                int j = ks - 64;
                valid = (j < NSTR) && ((45 + 46 * j) <= q - WWIN);
            }
            p[t][v] = valid ? c[v] * scale : NEGINF;
        }
    }

    // ---- softmax over keys: in-lane + one xor-16 combine ----
    float mx = NEGINF;
#pragma unroll
    for (int t = 0; t < 8; ++t)
#pragma unroll
        for (int v = 0; v < 8; ++v) mx = fmaxf(mx, p[t][v]);
    mx = fmaxf(mx, __shfl_xor(mx, 16, 32));
    float sum = 0.f;
#pragma unroll
    for (int t = 0; t < 8; ++t)
#pragma unroll
        for (int v = 0; v < 8; ++v) {
            float e = __expf(p[t][v] - mx);
            p[t][v] = e;
            sum += e;
        }
    sum += __shfl_xor(sum, 16, 32);
    float inv = 1.f / sum;
#pragma unroll
    for (int t = 0; t < 8; ++t)
#pragma unroll
        for (int v = 0; v < 8; ++v) p[t][v] *= inv;

    // ---- repack P^T (C-layout) into WMMA B fragments, 4 chunks of 32 keys --
    // B vgpr v of lane (n,g) needs keys k0=16g+2v, k0+1 (tile t=2c+g, local
    // index j=2v). Source register r=(2v)&7, source lane-group sg=(v>=4).
    FragBF pb[4];
#pragma unroll
    for (int cch = 0; cch < 4; ++cch) {
#pragma unroll
        for (int v = 0; v < 8; ++v) {
            const int r   = (2 * v) & 7;
            float y0  = p[2 * cch][r],     y0b = p[2 * cch][r + 1];
            float y1  = p[2 * cch + 1][r], y1b = p[2 * cch + 1][r + 1];
            float z0, z1;
            if (v < 4) {   // sg == 0
                float s1  = __shfl_xor(y1, 16, 32);
                float s1b = __shfl_xor(y1b, 16, 32);
                z0 = g ? s1 : y0;   z1 = g ? s1b : y0b;
            } else {       // sg == 1
                float s0  = __shfl_xor(y0, 16, 32);
                float s0b = __shfl_xor(y0b, 16, 32);
                z0 = g ? y1 : s0;   z1 = g ? y1b : s0b;
            }
            pb[cch].v[2 * v]     = (__bf16)z0;
            pb[cch].v[2 * v + 1] = (__bf16)z1;
        }
    }

    // ---- O^T = V^T * P^T over 4 d-tiles of 16 ----
#pragma unroll
    for (int dt = 0; dt < 4; ++dt) {
        v8f o = {};
#pragma unroll
        for (int cch = 0; cch < 4; ++cch) {
            FragBF a;   // V^T: row m = d_local (lane&15), K = key slots
#pragma unroll
            for (int e = 0; e < 16; ++e) {
                int kl   = g * 16 + e;
                int slot = cch * 32 + kl;
                int pos  = (slot < 64) ? (q0 - 48 + slot)
                                       : (45 + 46 * (slot - 64));
                int posc = pos < 0 ? 0 : (pos > S_LEN - 1 ? S_LEN - 1 : pos);
                a.v[e] = Vb[(long)posc * KVCOLS + hkv * HD + dt * 16 + n16];
            }
            o = __builtin_amdgcn_wmma_f32_16x16x32_bf16(false, a.v, false,
                                                        pb[cch].v, (short)0, o,
                                                        false, false);
        }
        // store: col n16 = query, rows = d_local 8g+v -> contiguous bf16 pairs
        __bf16* orow = Ob + (long)q * DMODEL + h * HD + dt * 16 + 8 * g;
#pragma unroll
        for (int v = 0; v < 4; ++v) {
            bf16x2 t2;
            t2[0] = (__bf16)o[2 * v];
            t2[1] = (__bf16)o[2 * v + 1];
            *(bf16x2*)(orow + 2 * v) = t2;
        }
    }
}

// ---------------------------------------------------------------------------
extern "C" void kernel_launch(void* const* d_in, const int* in_sizes, int n_in,
                              void* d_out, int out_size, void* d_ws, size_t ws_size,
                              hipStream_t stream) {
    (void)in_sizes; (void)n_in; (void)out_size; (void)ws_size;
    const float* x  = (const float*)d_in[0];
    const float* Wq = (const float*)d_in[1];
    const float* Wk = (const float*)d_in[2];
    const float* Wv = (const float*)d_in[3];
    const float* Wo = (const float*)d_in[4];

    char* ws = (char*)d_ws;
    size_t off = 0;
    auto alloc = [&](size_t bytes) -> void* {
        void* p = ws + off;
        off += (bytes + 255) & ~(size_t)255;
        return p;
    };
    __bf16* xb    = (__bf16*)alloc((size_t)S_LEN * DMODEL * 2);
    __bf16* wqb   = (__bf16*)alloc((size_t)DMODEL * DMODEL * 2);
    __bf16* wkb   = (__bf16*)alloc((size_t)KVCOLS * DMODEL * 2);
    __bf16* wvb   = (__bf16*)alloc((size_t)KVCOLS * DMODEL * 2);
    __bf16* wob   = (__bf16*)alloc((size_t)DMODEL * DMODEL * 2);
    __bf16* Qbuf  = (__bf16*)alloc((size_t)S_LEN * DMODEL * 2);
    __bf16* Kbuf  = (__bf16*)alloc((size_t)S_LEN * KVCOLS * 2);
    __bf16* Vbuf  = (__bf16*)alloc((size_t)S_LEN * KVCOLS * 2);
    __bf16* attnb = (__bf16*)alloc((size_t)S_LEN * DMODEL * 2);

    auto cvt = [&](const float* src, __bf16* dst, int n) {
        int n4 = n / 4;
        cvt_bf16_kernel<<<(n4 + 255) / 256, 256, 0, stream>>>(src, dst, n4);
    };
    cvt(x,  xb,  S_LEN * DMODEL);
    cvt(Wq, wqb, DMODEL * DMODEL);
    cvt(Wk, wkb, KVCOLS * DMODEL);
    cvt(Wv, wvb, KVCOLS * DMODEL);
    cvt(Wo, wob, DMODEL * DMODEL);

    auto gemm = [&](const __bf16* A, const __bf16* W, float* oF, __bf16* oB,
                    int M, int N, int K) {
        int waves  = (M / 64) * (N / 32);       // one 64x32 macro-tile / wave
        int blocks = (waves + 7) / 8;           // 8 waves / 256-thread block
        gemm_bf16_wmma<<<blocks, 256, 0, stream>>>(A, W, oF, oB, M, N, K);
    };
    gemm(xb, wqb, nullptr, Qbuf, S_LEN, DMODEL, DMODEL);
    gemm(xb, wkb, nullptr, Kbuf, S_LEN, KVCOLS, DMODEL);
    gemm(xb, wvb, nullptr, Vbuf, S_LEN, KVCOLS, DMODEL);

    // (S/16)*NH = 2048 waves -> 256 blocks of 8 waves
    attn_kernel<<<(S_LEN / 16) * NH / 8, 256, 0, stream>>>(Qbuf, Kbuf, Vbuf, attnb);

    gemm(attnb, wob, (float*)d_out, nullptr, S_LEN, DMODEL, DMODEL);
}
